// Int8GPTJAttention_2662879724430
// MI455X (gfx1250) — 8-bit matrix operand order repaired and checked by reading, compiled, not run on hardware
//
#include <hip/hip_runtime.h>
#include <hip/hip_bf16.h>
#include <math.h>

typedef int v8i __attribute__((ext_vector_type(8)));
typedef int v4i __attribute__((ext_vector_type(4)));
typedef int gv4 __attribute__((vector_size(16)));

#define T_SEQ 2048
#define E_DIM 4096
#define H_NUM 16
#define D_HEAD 256

__device__ __forceinline__ v8i zero8() {
    v8i z;
#pragma unroll
    for (int i = 0; i < 8; ++i) z[i] = 0;
    return z;
}

typedef __attribute__((address_space(1))) gv4* g_v4p;
typedef __attribute__((address_space(3))) gv4* l_v4p;

__device__ __forceinline__ void async_cp_b128(void* lds, const void* g) {
#if defined(__AMDGCN__) && __has_builtin(__builtin_amdgcn_global_load_async_to_lds_b128)
    __builtin_amdgcn_global_load_async_to_lds_b128((g_v4p)g, (l_v4p)lds, 0, 0);
#else
    unsigned lo = (unsigned)(size_t)((__attribute__((address_space(3))) void*)lds);
    asm volatile("global_load_async_to_lds_b128 %0, %1, off"
                 :: "v"(lo), "v"((unsigned long long)(size_t)g) : "memory");
#endif
}

#if defined(__AMDGCN__) && __has_builtin(__builtin_amdgcn_s_wait_asynccnt)
#define WAIT_ASYNC(n) __builtin_amdgcn_s_wait_asynccnt(n)
#else
#define WAIT_ASYNC(n) asm volatile("s_wait_asynccnt %0" :: "i"(n) : "memory")
#endif

__device__ __forceinline__ v8i load_a8(const signed char* base, int stride, int lane) {
    const int ln = lane & 15, half = (lane >> 4) & 1;
    const signed char* p = base + ln * stride + 8 * half;
    v8i a;
#pragma unroll
    for (int pr = 0; pr < 4; ++pr) {
        a[2 * pr]     = *(const int*)(p + 16 * pr);
        a[2 * pr + 1] = *(const int*)(p + 16 * pr + 4);
    }
    return a;
}

__device__ __forceinline__ v8i load_b8(const signed char* base, int stride, int lane) {
    const int ln = lane & 15, half = (lane >> 4) & 1;
    const signed char* p = base + ln * stride + 8 * half;
    v8i b;
#pragma unroll
    for (int pr = 0; pr < 4; ++pr) {
        b[2 * pr]     = *(const int*)(p + 16 * pr);
        b[2 * pr + 1] = *(const int*)(p + 16 * pr + 4);
    }
    return b;
}

__device__ __forceinline__ int clip8(int r) {
    return r < -128 ? -128 : (r > 127 ? 127 : r);
}

__global__ void pack_i8_kernel(const int* __restrict__ in, signed char* __restrict__ out, int n) {
    int i = blockIdx.x * blockDim.x + threadIdx.x;
    if (i < n) out[i] = (signed char)in[i];
}

__device__ __forceinline__ void gemm128_accum(const signed char* __restrict__ A,
                                              const signed char* __restrict__ B,
                                              int t0, int o0, int tid, v8i acc[8]) {
    __shared__ signed char Abuf[2][128 * 64];
    __shared__ signed char Bbuf[2][128 * 64];

    const int lane = tid & 31;
    const int ln = lane & 15, half = (lane >> 4) & 1;
    const int wid = tid >> 5;
    const int wm = wid >> 2, wn = wid & 3;
    const int r = tid >> 2, seg = tid & 3;

    const signed char* Ag = A + (size_t)(t0 + r) * E_DIM + seg * 16;
    const signed char* Bg = B + (size_t)(o0 + r) * E_DIM + seg * 16;

    auto issue = [&](int kb) {
        const int buf = (kb >> 6) & 1;
        signed char* al = &Abuf[buf][r * 64 + seg * 16];
        signed char* bl = &Bbuf[buf][r * 64 + seg * 16];
        async_cp_b128(al,            Ag + kb);
        async_cp_b128(al + 64 * 64,  Ag + (size_t)64 * E_DIM + kb);
        async_cp_b128(bl,            Bg + kb);
        async_cp_b128(bl + 64 * 64,  Bg + (size_t)64 * E_DIM + kb);
    };

#pragma unroll
    for (int i = 0; i < 8; ++i) acc[i] = zero8();

    issue(0);
    for (int kb = 0; kb < E_DIM; kb += 64) {
        if (kb + 64 < E_DIM) {
            issue(kb + 64);
            WAIT_ASYNC(4);
        } else {
            WAIT_ASYNC(0);
        }
        __syncthreads();

        const signed char* Al = Abuf[(kb >> 6) & 1];
        const signed char* Bl = Bbuf[(kb >> 6) & 1];

        v8i af[4];
#pragma unroll
        for (int f = 0; f < 4; ++f) {
            const signed char* p = Al + (wm * 64 + f * 16 + ln) * 64 + 8 * half;
#pragma unroll
            for (int pr = 0; pr < 4; ++pr) {
                af[f][2 * pr]     = *(const int*)(p + 16 * pr);
                af[f][2 * pr + 1] = *(const int*)(p + 16 * pr + 4);
            }
        }
        v8i bf[2];
#pragma unroll
        for (int g = 0; g < 2; ++g) {
            const signed char* p = Bl + (wn * 32 + g * 16 + ln) * 64 + 8 * half;
#pragma unroll
            for (int pr = 0; pr < 4; ++pr) {
                bf[g][2 * pr]     = *(const int*)(p + 16 * pr);
                bf[g][2 * pr + 1] = *(const int*)(p + 16 * pr + 4);
            }
        }
#pragma unroll
        for (int f = 0; f < 4; ++f)
#pragma unroll
            for (int g = 0; g < 2; ++g)
                acc[f * 2 + g] = __builtin_amdgcn_wmma_i32_16x16x64_iu8(
                    true, af[f], true, bf[g], acc[f * 2 + g], false, false);

        __syncthreads();
    }
}

__global__ void qkv_gemm_kernel(const signed char* __restrict__ x8,
                                const signed char* __restrict__ W8,
                                const signed char* __restrict__ b8,
                                const float* __restrict__ alpha_p,
                                const float* __restrict__ beta_p,
                                signed char* __restrict__ out, int vmode) {
    const int tid = threadIdx.x;
    const int tm = blockIdx.x >> 5, tn = blockIdx.x & 31;
    const int t0 = tm * 128, o0 = tn * 128;
    const float alpha = *alpha_p, beta = *beta_p;

    v8i acc[8];
    gemm128_accum(x8, W8, t0, o0, tid, acc);

    const int lane = tid & 31, wid = tid >> 5;
    const int ln = lane & 15, half = (lane >> 4) & 1;
    const int mw = t0 + (wid >> 2) * 64;
    const int nw = o0 + (wid & 3) * 32;
#pragma unroll
    for (int f = 0; f < 4; ++f) {
#pragma unroll
        for (int g = 0; g < 2; ++g) {
            const v8i a = acc[f * 2 + g];
            const int o = nw + g * 16 + ln;
            const int h = o >> 8, d = o & 255;
            const float bias = (float)b8[o];
#pragma unroll
            for (int j = 0; j < 8; ++j) {
                const int t = mw + f * 16 + j + 8 * half;
                const float y = alpha * (float)a[j] + beta * bias;
                const int rq = clip8((int)rintf(y));
                size_t addr;
                if (vmode) addr = ((size_t)(h * D_HEAD + d) << 11) + (size_t)t;
                else       addr = ((size_t)(h * T_SEQ + t) << 8) + (size_t)d;
                out[addr] = (signed char)rq;
            }
        }
    }
}

__global__ void rotary_kernel(signed char* __restrict__ q, signed char* __restrict__ k) {
    const int idx = blockIdx.x * blockDim.x + threadIdx.x;
    if (idx >= H_NUM * T_SEQ * 32) return;
    const int j = idx & 31;
    const int t = (idx >> 5) & (T_SEQ - 1);
    const int h = idx >> 16;
    const float f = expf(-((float)(2 * j) / 64.0f) * 9.210340371976184f);
    const float ang = (float)t * f;
    const float s = sinf(ang), c = cosf(ang);
    const size_t base = ((size_t)(h * T_SEQ + t) << 8) + 2 * j;
#pragma unroll
    for (int which = 0; which < 2; ++which) {
        signed char* buf = which ? k : q;
        const float x1 = (float)buf[base], x2 = (float)buf[base + 1];
        const float y1 = x1 * c - x2 * s;
        const float y2 = x2 * c + x1 * s;
        buf[base]     = (signed char)clip8((int)truncf(y1));
        buf[base + 1] = (signed char)clip8((int)truncf(y2));
    }
}

__global__ void attn_kernel(const signed char* __restrict__ q,
                            const signed char* __restrict__ k,
                            const signed char* __restrict__ vT,
                            const float* __restrict__ alpha_qk_p,
                            const float* __restrict__ alpha_pv_p,
                            signed char* __restrict__ attn8) {
    extern __shared__ char smem[];
    float*       scores = (float*)smem;
    signed char* probs  = (signed char*)(smem + 16 * T_SEQ * 4);
    float*       red    = (float*)(smem + 16 * T_SEQ * 4 + 16 * T_SEQ);

    const int h = blockIdx.y, it = blockIdx.x;
    const int t0 = it * 16;
    const int ncols = (it + 1) * 16;
    const int ncols64 = (ncols + 63) & ~63;
    const int tid = threadIdx.x, lane = tid & 31, wid = tid >> 5;
    const int ln = lane & 15, half = lane >> 4;
    const float aqk = (*alpha_qk_p) * (1.0f / 16.0f);
    const float apv = *alpha_pv_p;

    const signed char* qbase = q + (((size_t)(h * T_SEQ + t0)) << 8);
    v8i aq[4];
#pragma unroll
    for (int c = 0; c < 4; ++c) aq[c] = load_a8(qbase + 64 * c, D_HEAD, lane);

    for (int js = wid; js <= it; js += 4) {
        const signed char* kbase = k + (((size_t)(h * T_SEQ + js * 16)) << 8);
        v8i acc = zero8();
#pragma unroll
        for (int c = 0; c < 4; ++c) {
            v8i b = load_b8(kbase + 64 * c, D_HEAD, lane);
            acc = __builtin_amdgcn_wmma_i32_16x16x64_iu8(true, aq[c], true, b, acc, false, false);
        }
        const int s = js * 16 + ln;
#pragma unroll
        for (int j = 0; j < 8; ++j) {
            const int m = j + 8 * half;
            scores[m * T_SEQ + s] = (s <= t0 + m) ? (float)acc[j] * aqk : -3.0e38f;
        }
    }
    __syncthreads();

    const int r = tid & 15, part = tid >> 4;
    float pmax = -3.0e38f;
    for (int c2 = part; c2 < ncols; c2 += 8) pmax = fmaxf(pmax, scores[r * T_SEQ + c2]);
    red[r * 8 + part] = pmax;
    __syncthreads();
    float rowmax = red[r * 8];
#pragma unroll
    for (int i = 1; i < 8; ++i) rowmax = fmaxf(rowmax, red[r * 8 + i]);
    __syncthreads();
    float psum = 0.0f;
    for (int c2 = part; c2 < ncols; c2 += 8) psum += __expf(scores[r * T_SEQ + c2] - rowmax);
    red[r * 8 + part] = psum;
    __syncthreads();
    float denom = 0.0f;
#pragma unroll
    for (int i = 0; i < 8; ++i) denom += red[r * 8 + i];
    const float qs = 127.0f / denom;
    for (int c2 = part; c2 < ncols; c2 += 8) {
        const float e = __expf(scores[r * T_SEQ + c2] - rowmax);
        probs[r * T_SEQ + c2] = (signed char)((int)rintf(e * qs));
    }
    for (int c2 = ncols + part; c2 < ncols64; c2 += 8) probs[r * T_SEQ + c2] = 0;
    __syncthreads();

    for (int dt = wid; dt < 16; dt += 4) {
        const int d0 = dt * 16;
        v8i acc = zero8();
        for (int sc = 0; sc < ncols64; sc += 64) {
            v8i a;
            const signed char* ap = probs + ln * T_SEQ + sc + 8 * half;
#pragma unroll
            for (int pr = 0; pr < 4; ++pr) {
                a[2 * pr]     = *(const int*)(ap + 16 * pr);
                a[2 * pr + 1] = *(const int*)(ap + 16 * pr + 4);
            }
            const signed char* bp = vT + (((size_t)(h * D_HEAD + d0)) << 11) + sc;
            v8i b = load_b8(bp, T_SEQ, lane);
            acc = __builtin_amdgcn_wmma_i32_16x16x64_iu8(true, a, true, b, acc, false, false);
        }
#pragma unroll
        for (int j = 0; j < 8; ++j) {
            const int m = j + 8 * half;
            const int val = clip8((int)rintf(apv * (float)acc[j]));
            attn8[(((size_t)(t0 + m)) << 12) + (size_t)(h * D_HEAD + d0 + ln)] = (signed char)val;
        }
    }
}

__global__ void out_gemm_kernel(const signed char* __restrict__ a8,
                                const signed char* __restrict__ W8,
                                const float* __restrict__ bias,
                                const float* __restrict__ alpha_p,
                                float* __restrict__ out) {
    const int tid = threadIdx.x;
    const int tm = blockIdx.x >> 5, tn = blockIdx.x & 31;
    const int t0 = tm * 128, o0 = tn * 128;
    const float alpha = *alpha_p;

    v8i acc[8];
    gemm128_accum(a8, W8, t0, o0, tid, acc);

    const int lane = tid & 31, wid = tid >> 5;
    const int ln = lane & 15, half = (lane >> 4) & 1;
    const int mw = t0 + (wid >> 2) * 64;
    const int nw = o0 + (wid & 3) * 32;
#pragma unroll
    for (int f = 0; f < 4; ++f) {
#pragma unroll
        for (int g = 0; g < 2; ++g) {
            const v8i a = acc[f * 2 + g];
            const int o = nw + g * 16 + ln;
            const float bn = bias[o];
#pragma unroll
            for (int j = 0; j < 8; ++j) {
                const int t = mw + f * 16 + j + 8 * half;
                out[(size_t)t * E_DIM + (size_t)o] = alpha * (float)a[j] + bn;
            }
        }
    }
}

extern "C" void kernel_launch(void* const* d_in, const int* in_sizes, int n_in,
                              void* d_out, int out_size, void* d_ws, size_t ws_size,
                              hipStream_t stream) {
    const int*   hs    = (const int*)d_in[0];
    const int*   Wq    = (const int*)d_in[1];
    const int*   bq    = (const int*)d_in[2];
    const int*   Wk    = (const int*)d_in[3];
    const int*   bk    = (const int*)d_in[4];
    const int*   Wv    = (const int*)d_in[5];
    const int*   bv    = (const int*)d_in[6];
    const int*   Wo    = (const int*)d_in[7];
    const float* b_out = (const float*)d_in[8];
    const float* alpha_q   = (const float*)d_in[9];
    const float* beta_q    = (const float*)d_in[10];
    const float* alpha_k   = (const float*)d_in[11];
    const float* beta_k    = (const float*)d_in[12];
    const float* alpha_v   = (const float*)d_in[13];
    const float* beta_v    = (const float*)d_in[14];
    const float* alpha_qk  = (const float*)d_in[15];
    const float* alpha_pv  = (const float*)d_in[16];
    const float* alpha_out = (const float*)d_in[17];

    char* ws = (char*)d_ws;
    size_t off = 0;
    auto carve = [&](size_t sz) { size_t o = off; off += (sz + 4095) & ~(size_t)4095; return o; };
    const size_t N_HS = (size_t)T_SEQ * E_DIM;
    const size_t N_W  = (size_t)E_DIM * E_DIM;
    signed char* hs8   = (signed char*)(ws + carve(N_HS));
    signed char* wq8   = (signed char*)(ws + carve(N_W));
    signed char* wk8   = (signed char*)(ws + carve(N_W));
    signed char* wv8   = (signed char*)(ws + carve(N_W));
    signed char* wo8   = (signed char*)(ws + carve(N_W));
    signed char* bq8   = (signed char*)(ws + carve(E_DIM));
    signed char* bk8   = (signed char*)(ws + carve(E_DIM));
    signed char* bv8   = (signed char*)(ws + carve(E_DIM));
    signed char* q_hm  = (signed char*)(ws + carve(N_HS));
    signed char* k_hm  = (signed char*)(ws + carve(N_HS));
    signed char* v_t   = (signed char*)(ws + carve(N_HS));
    signed char* attn8 = (signed char*)(ws + carve(N_HS));

    auto packN = [&](const int* src, signed char* dst, size_t n) {
        pack_i8_kernel<<<(unsigned)((n + 255) / 256), 256, 0, stream>>>(src, dst, (int)n);
    };
    packN(hs, hs8, N_HS);
    packN(Wq, wq8, N_W);
    packN(Wk, wk8, N_W);
    packN(Wv, wv8, N_W);
    packN(Wo, wo8, N_W);
    packN(bq, bq8, E_DIM);
    packN(bk, bk8, E_DIM);
    packN(bv, bv8, E_DIM);

    qkv_gemm_kernel<<<512, 256, 0, stream>>>(hs8, wq8, bq8, alpha_q, beta_q, q_hm, 0);
    qkv_gemm_kernel<<<512, 256, 0, stream>>>(hs8, wk8, bk8, alpha_k, beta_k, k_hm, 0);
    qkv_gemm_kernel<<<512, 256, 0, stream>>>(hs8, wv8, bv8, alpha_v, beta_v, v_t, 1);

    rotary_kernel<<<(H_NUM * T_SEQ * 32) / 256, 256, 0, stream>>>(q_hm, k_hm);

    const size_t smem = 16 * T_SEQ * 4 + 16 * T_SEQ + 16 * 8 * 4;
    dim3 agrid(T_SEQ / 16, H_NUM);
    attn_kernel<<<agrid, 128, smem, stream>>>(q_hm, k_hm, v_t, alpha_qk, alpha_pv, attn8);

    out_gemm_kernel<<<512, 256, 0, stream>>>(attn8, wo8, b_out, alpha_out, (float*)d_out);
}
